// Generator_1331439862244
// MI455X (gfx1250) — matrix operand order repaired and checked by reading, not compiled, not run on hardware
//
#include <hip/hip_runtime.h>
#include <hip/hip_bf16.h>
#include <math.h>

typedef _Float16 half8  __attribute__((ext_vector_type(8)));
typedef _Float16 half16 __attribute__((ext_vector_type(16)));
typedef float    float8 __attribute__((ext_vector_type(8)));

#define B_    8
#define CIN   256
#define COUT  256
#define H_    64
#define W_    64
#define HO    62
#define WO    62
#define K9    9
#define KTOT  2304      // CIN*9
#define SDIM  512

// ---------------------------------------------------------------------------
// Kernel 1: style projection  s[b][ci] = style[b,:]·mod_w[ci,:]/sqrt(512)+mod_b
// ---------------------------------------------------------------------------
__global__ void k_style(const float* __restrict__ style, const float* __restrict__ mod_w,
                        const float* __restrict__ mod_b, float* __restrict__ s_out) {
  int tid = blockIdx.x * blockDim.x + threadIdx.x;   // 0..2047
  int b = tid >> 8, ci = tid & 255;
  const float invs = 0.04419417382415922f;           // 1/sqrt(512)
  const float* st = style + b * SDIM;
  const float* mw = mod_w + ci * SDIM;
  float acc = 0.f;
  for (int d = 0; d < SDIM; ++d) acc += st[d] * mw[d];
  s_out[tid] = acc * invs + mod_b[ci];
}

// ---------------------------------------------------------------------------
// Kernel 2: modulate + demodulate weights -> f16, layout [b][co][ci*9+kk]
// ---------------------------------------------------------------------------
__global__ void k_modweight(const float* __restrict__ cw, const float* __restrict__ s_buf,
                            _Float16* __restrict__ w16) {
  int b  = blockIdx.y;
  int co = blockIdx.x;
  int t  = threadIdx.x;                              // = ci
  const float scale = 1.0f / 48.0f;                  // 1/sqrt(256*9)
  float sv = s_buf[b * CIN + t];
  const float* wrow = cw + (co * CIN + t) * K9;
  float vals[9], ss = 0.f;
  #pragma unroll
  for (int k = 0; k < 9; ++k) { float v = scale * wrow[k] * sv; vals[k] = v; ss += v * v; }
  __shared__ float red[256];
  __shared__ float dem;
  red[t] = ss;
  __syncthreads();
  for (int off = 128; off > 0; off >>= 1) {
    if (t < off) red[t] += red[t + off];
    __syncthreads();
  }
  if (t == 0) dem = rsqrtf(red[0] + 1e-8f) * (1.0f / (1.0f + 1e-8f));
  __syncthreads();
  float d = dem;
  _Float16* out = w16 + (size_t)(b * COUT + co) * KTOT + t * K9;
  #pragma unroll
  for (int k = 0; k < 9; ++k) out[k] = (_Float16)(vals[k] * d);
}

// ---------------------------------------------------------------------------
// Kernel 3: implicit-GEMM modulated conv via v_wmma_f32_16x16x32_f16
// per batch GEMM: M=co(256), N=p(62x62), K=ci*9(2304)
// block: 256 thr = 8 wave32; block tile 64co x (32x * 2y); K chunk 64 via LDS
// wave (wc,wy): A frag 16co shared across two 16x sub-tiles -> 4 WMMA/stage
// ---------------------------------------------------------------------------
#define KC      64
#define APITCH  72      // halfs (144B, 16B aligned)
#define BPITCH  72

__global__ void __launch_bounds__(256)
k_conv_wmma(const float* __restrict__ xin, const _Float16* __restrict__ w16,
            const float* __restrict__ bias, float* __restrict__ conv_out) {
  __shared__ _Float16 sA[64 * APITCH];   // 64 co x 64 k
  __shared__ _Float16 sB[64 * BPITCH];   // (2y x 32x) x 64 k

  const int xb   = blockIdx.x;        // 0..1   (x strip of 32)
  const int yb   = blockIdx.y;        // 0..30  (pair of y rows)
  const int cob  = blockIdx.z & 3;    // co block of 64
  const int b    = blockIdx.z >> 2;
  const int t    = threadIdx.x;
  const int lane = t & 31;
  const int wv   = t >> 5;            // wave 0..7
  const int wc   = wv & 3;            // co sub-tile (16 each)
  const int wy   = wv >> 2;           // which y row (0/1)
  const int half = lane >> 4;         // lane 16..31 -> 1
  const int m    = lane & 15;

  const _Float16* gA = w16 + (size_t)(b * COUT + cob * 64) * KTOT;
  const float*    gX = xin + (size_t)b * CIN * H_ * W_;
  const int x_base = xb * 32;
  const int y0     = yb * 2;

  // per-thread fixed B-staging coordinates (invariant across k stages)
  const int xi0 = (t & 7) * 4;        // group of 4 x
  const int py  = (t >> 3) & 1;       // y row within tile
  const int kl0 = t >> 4;             // 0..15; handles kl0 + 16u, u=0..3

  // incremental (ci,kk) trackers for kl = kl0 + 16u ; k advances by 64/stage
  int ciu[4], kku[4];
  {
    int c0 = kl0 / 9, q0 = kl0 % 9;   // one-time const division
    #pragma unroll
    for (int u = 0; u < 4; ++u) {
      ciu[u] = c0; kku[u] = q0;
      c0 += 1; q0 += 7;               // advance kl by 16 = 1*9 + 7
      if (q0 >= 9) { q0 -= 9; c0 += 1; }
    }
  }

  float8 acc0 = {};
  float8 acc1 = {};

  for (int k0 = 0; k0 < KTOT; k0 += KC) {
    // ---- stage A: 64 co rows x 64 halfs (contiguous b128 copies) ----
    #pragma unroll
    for (int u = 0; u < 2; ++u) {
      int lin = t + 256 * u;          // 0..511
      int row = lin >> 3;             // 0..63
      int seg = lin & 7;              // 8 x 16B per row
      uint4 v = *reinterpret_cast<const uint4*>(gA + row * KTOT + k0 + seg * 8);
      *reinterpret_cast<uint4*>(&sA[row * APITCH + seg * 8]) = v;
      if (k0 + KC < KTOT)             // prefetch next weight chunk into caches
        __builtin_prefetch(gA + row * KTOT + (k0 + KC) + seg * 8, 0, 3);
    }
    // ---- stage B: im2col 64k x (2y x 32x), stored [p][k] for contig frags
    #pragma unroll
    for (int u = 0; u < 4; ++u) {
      int kl = kl0 + 16 * u;          // 0..63
      int ci = ciu[u], kk = kku[u];
      int ky = kk / 3, kx = kk % 3;   // const-divisor, mul-based
      const float* src = gX + (ci * H_ + (y0 + py + ky)) * W_;
      #pragma unroll
      for (int j = 0; j < 4; ++j) {
        int xi = x_base + xi0 + j + kx;
        float v = (xi < W_) ? src[xi] : 0.f;
        sB[(py * 32 + xi0 + j) * BPITCH + kl] = (_Float16)v;
      }
      // advance this tracker by 64 = 7*9 + 1 for the next stage
      kku[u] += 1; ciu[u] += 7;
      if (kku[u] >= 9) { kku[u] -= 9; ciu[u] += 1; }
    }
    __syncthreads();
    // ---- four WMMAs per stage: 2 K-chunks x 2 x-sub-tiles ----
    #pragma unroll
    for (int c = 0; c < 2; ++c) {
      const int ka = c * 32;
      // A frag: lane half h: K = h*16 + {0..15}
      half8 a_lo = *reinterpret_cast<const half8*>(&sA[(wc * 16 + m) * APITCH + ka + half * 16]);
      half8 a_hi = *reinterpret_cast<const half8*>(&sA[(wc * 16 + m) * APITCH + ka + half * 16 + 8]);
      half16 av;
      #pragma unroll
      for (int j = 0; j < 8; ++j) { av[j] = a_lo[j]; av[8 + j] = a_hi[j]; }
      #pragma unroll
      for (int s = 0; s < 2; ++s) {
        // B frag: lane half h: K = h*16 + {0..15}, contiguous in [p][k]
        const int prow = (wy * 32 + s * 16 + m) * BPITCH + ka + half * 16;
        half8 b_lo = *reinterpret_cast<const half8*>(&sB[prow]);
        half8 b_hi = *reinterpret_cast<const half8*>(&sB[prow + 8]);
        half16 bv;
        #pragma unroll
        for (int j = 0; j < 8; ++j) { bv[j] = b_lo[j]; bv[8 + j] = b_hi[j]; }
        if (s == 0)
          acc0 = __builtin_amdgcn_wmma_f32_16x16x32_f16(false, av, false, bv,
                                                        (short)0, acc0, false, false);
        else
          acc1 = __builtin_amdgcn_wmma_f32_16x16x32_f16(false, av, false, bv,
                                                        (short)0, acc1, false, false);
      }
    }
    __syncthreads();
  }

  // ---- epilogue: C layout lanes0-15 M=r, lanes16-31 M=8+r ----
  const int y_out = y0 + wy;
  #pragma unroll
  for (int s = 0; s < 2; ++s) {
    const int x_out = x_base + s * 16 + m;
    if (x_out < WO) {
      #pragma unroll
      for (int r = 0; r < 8; ++r) {
        int co = cob * 64 + wc * 16 + half * 8 + r;
        float v = (s == 0 ? acc0[r] : acc1[r]) + bias[co];
        conv_out[((size_t)(b * COUT + co) * HO + y_out) * WO + x_out] = v;
      }
    }
  }
}

// ---------------------------------------------------------------------------
// Kernels 4-7: separable upfirdn passes (bandwidth-bound, 1 thread/output)
// ---------------------------------------------------------------------------
__global__ void k_upx(const float* __restrict__ in, const float* __restrict__ filt,
                      float* __restrict__ out) {          // [2048][62][62] -> [2048][62][128]
  int idx = blockIdx.x * 256 + threadIdx.x;
  int X = idx & 127;
  int r = idx >> 7;
  int y = r % HO;
  int bc = r / HO;
  const float* src = in + ((size_t)bc * HO + y) * WO;
  int jlo = (X >= 8) ? ((X - 7) >> 1) : 0;
  int jhi = (X + 3) >> 1; if (jhi > WO - 1) jhi = WO - 1;
  float acc = 0.f;
  for (int j = jlo; j <= jhi; ++j) acc += src[j] * filt[X - 2 * j + 3];
  out[idx] = acc * 2.0f;                                  // uf = up_filter * UP
}

__global__ void k_upy(const float* __restrict__ in, const float* __restrict__ filt,
                      float* __restrict__ out) {          // -> [2048][128][128] + leaky*sqrt2
  int idx = blockIdx.x * 256 + threadIdx.x;
  int X = idx & 127;
  int r = idx >> 7;
  int Y = r & 127;
  int bc = r >> 7;
  const float* src = in + (size_t)bc * HO * 128 + X;
  int jlo = (Y >= 8) ? ((Y - 7) >> 1) : 0;
  int jhi = (Y + 3) >> 1; if (jhi > HO - 1) jhi = HO - 1;
  float acc = 0.f;
  for (int j = jlo; j <= jhi; ++j) acc += src[j * 128] * filt[Y - 2 * j + 3];
  acc *= 2.0f;
  acc = (acc >= 0.f) ? acc : 0.2f * acc;
  out[idx] = acc * 1.4142135623730951f;
}

__global__ void k_downx(const float* __restrict__ in, const float* __restrict__ filt,
                        float* __restrict__ out) {        // -> [2048][128][64]
  int idx = blockIdx.x * 256 + threadIdx.x;
  int X = idx & 63;
  int r = idx >> 6;
  int y = r & 127;
  int bc = r >> 7;
  const float* src = in + ((size_t)bc * 128 + y) * 128;
  int jlo = 2 * X - 5; if (jlo < 0) jlo = 0;
  int jhi = 2 * X + 6; if (jhi > 127) jhi = 127;
  float acc = 0.f;
  for (int j = jlo; j <= jhi; ++j) acc += src[j] * filt[2 * X + 6 - j];
  out[idx] = acc;
}

__global__ void k_downy(const float* __restrict__ in, const float* __restrict__ filt,
                        float* __restrict__ out) {        // -> [2048][64][64]
  int idx = blockIdx.x * 256 + threadIdx.x;
  int x = idx & 63;
  int r = idx >> 6;
  int Y = r & 63;
  int bc = r >> 6;
  const float* src = in + (size_t)bc * 128 * 64 + x;
  int jlo = 2 * Y - 5; if (jlo < 0) jlo = 0;
  int jhi = 2 * Y + 6; if (jhi > 127) jhi = 127;
  float acc = 0.f;
  for (int j = jlo; j <= jhi; ++j) acc += src[j * 64] * filt[2 * Y + 6 - j];
  out[idx] = acc;
}

// ---------------------------------------------------------------------------
// Launch
// ---------------------------------------------------------------------------
extern "C" void kernel_launch(void* const* d_in, const int* in_sizes, int n_in,
                              void* d_out, int out_size, void* d_ws, size_t ws_size,
                              hipStream_t stream) {
  const float* input  = (const float*)d_in[0];
  const float* style  = (const float*)d_in[1];
  const float* conv_w = (const float*)d_in[2];
  const float* mod_w  = (const float*)d_in[3];
  const float* mod_b  = (const float*)d_in[4];
  const float* bias   = (const float*)d_in[5];
  const float* up_f   = (const float*)d_in[6];
  const float* down_f = (const float*)d_in[7];
  float* out = (float*)d_out;

  char* ws = (char*)d_ws;
  _Float16* w16  = (_Float16*)(ws + 0);          //  9,437,184 B : f16 weights
  float*    sbuf = (float*)(ws + 9437184);       //      8,192 B : s[b][ci]
  float*    t0   = (float*)(ws + 9445376);       // 31,490,048 B : conv out (8,256,62,62)
  float*    t1   = (float*)(ws + 40935424);      // 65,011,712 B : up-x   (8,256,62,128)
  float*    t2   = (float*)(ws + 105947136);     //134,217,728 B : up-y   (8,256,128,128)
  float*    t3   = t0;                           // down-x reuses t0+t1 region (67,108,864 B)

  k_style<<<8, 256, 0, stream>>>(style, mod_w, mod_b, sbuf);
  dim3 gw(256, 8);
  k_modweight<<<gw, 256, 0, stream>>>(conv_w, sbuf, w16);
  dim3 gc(2, 31, 32);                            // x strips(32), y pairs, b*co_blocks
  k_conv_wmma<<<gc, 256, 0, stream>>>(input, w16, bias, t0);
  k_upx  <<<16252928 / 256, 256, 0, stream>>>(t0, up_f, t1);
  k_upy  <<<33554432 / 256, 256, 0, stream>>>(t1, up_f, t2);
  k_downx<<<16777216 / 256, 256, 0, stream>>>(t2, down_f, t3);
  k_downy<<< 8388608 / 256, 256, 0, stream>>>(t3, down_f, out);
}
